// AnimalGuy_6502580486537
// MI455X (gfx1250) — matrix operand order repaired and checked by reading, not compiled, not run on hardware
//
#include <hip/hip_runtime.h>
#include <cstdint>
#include <cstddef>

#define D_MODEL 1024
#define NHEAD   16
#define DHEAD   64
#define NLAYER  4
#define KSEL    256
#define MMEM    100000
#define SEQ     514          /* 2*K + 2 */
#define RBINS   131072       /* rank histogram bins: key >> 15 */

typedef __attribute__((ext_vector_type(16))) __bf16         bf16x16;
typedef __attribute__((ext_vector_type(16))) unsigned short u16x16;
typedef __attribute__((ext_vector_type(8)))  float          f32x8;

#if __has_builtin(__builtin_amdgcn_tensor_load_to_lds)
#define GEMM_TDM 1
typedef unsigned int u32x4v __attribute__((ext_vector_type(4)));
typedef int          i32x8v __attribute__((ext_vector_type(8)));
typedef int          i32x4v __attribute__((ext_vector_type(4)));
#endif

__device__ __forceinline__ unsigned short f2bf(float f) {
  unsigned u = __float_as_uint(f);
  u += 0x7fffu + ((u >> 16) & 1u);          // round-to-nearest-even
  return (unsigned short)(u >> 16);
}
__device__ __forceinline__ uint2 pack4bf(float4 v) {
  uint2 r;
  r.x = (unsigned)f2bf(v.x) | ((unsigned)f2bf(v.y) << 16);
  r.y = (unsigned)f2bf(v.z) | ((unsigned)f2bf(v.w) << 16);
  return r;
}
__device__ __forceinline__ unsigned sortable_key(float f) {
  unsigned u = __float_as_uint(f);
  return (u & 0x80000000u) ? ~u : (u | 0x80000000u);
}
__device__ __forceinline__ float gelu_f(float x) {
  return 0.5f * x * (1.0f + erff(x * 0.7071067811865475f));
}

// ---------------------------------------------------------------- utilities
__global__ void __launch_bounds__(256) k_zero_u32(unsigned* p, int n) {
  int i = blockIdx.x * 256 + threadIdx.x;
  if (i < n) p[i] = 0u;
}

__global__ void __launch_bounds__(256) k_xnorm(const float* __restrict__ xv,
                                               float* scalf) {
  __shared__ float red[9];
  float s = 0.f;
  for (int i = threadIdx.x; i < D_MODEL; i += 256) { float v = xv[i]; s += v * v; }
#pragma unroll
  for (int o = 16; o > 0; o >>= 1) s += __shfl_xor(s, o, 32);
  if ((threadIdx.x & 31) == 0) red[threadIdx.x >> 5] = s;
  __syncthreads();
  if (threadIdx.x == 0) {
    float t = 0.f;
    for (int i = 0; i < 8; ++i) t += red[i];
    scalf[0] = sqrtf(t);
  }
}

// ---------------------------------------------------------------- similarity
__global__ void __launch_bounds__(256) k_sims(const float* __restrict__ mv,
                                              const float* __restrict__ xv,
                                              const float* __restrict__ scalf,
                                              float* __restrict__ sims) {
  int wave = threadIdx.x >> 5, lane = threadIdx.x & 31;
  long long row = (long long)blockIdx.x * 8 + wave;
  if (row >= MMEM) return;
  const float4* r4 = (const float4*)(mv + (size_t)row * D_MODEL);
  const float4* x4 = (const float4*)xv;
  if (row + 8 < MMEM)
    __builtin_prefetch(mv + (size_t)(row + 8) * D_MODEL + lane * 32, 0, 0);
  float dot = 0.f, ss = 0.f;
#pragma unroll
  for (int i = 0; i < 8; ++i) {
    float4 a = r4[lane + i * 32];
    float4 b = x4[lane + i * 32];
    dot += a.x * b.x + a.y * b.y + a.z * b.z + a.w * b.w;
    ss  += a.x * a.x + a.y * a.y + a.z * a.z + a.w * a.w;
  }
#pragma unroll
  for (int o = 16; o > 0; o >>= 1) {
    dot += __shfl_xor(dot, o, 32);
    ss  += __shfl_xor(ss,  o, 32);
  }
  if (lane == 0) sims[row] = dot / fmaxf(sqrtf(ss) * scalf[0], 1e-8f);
}

// ---------------------------------------------------------------- rank scores
__global__ void __launch_bounds__(256) k_keys(const float* __restrict__ v,
                                              unsigned* __restrict__ key, int n) {
  int i = blockIdx.x * 256 + threadIdx.x;
  if (i < n) key[i] = sortable_key(v[i]);
}

__global__ void __launch_bounds__(256) k_hist15(const unsigned* __restrict__ key,
                                                unsigned* __restrict__ hist, int n) {
  int i = blockIdx.x * 256 + threadIdx.x;
  if (i < n) atomicAdd(&hist[key[i] >> 15], 1u);
}

__global__ void __launch_bounds__(1024) k_prefix(const unsigned* __restrict__ hist,
                                                 unsigned* __restrict__ pref, int nbins) {
  __shared__ unsigned tmp[1024];
  __shared__ unsigned carry;
  if (threadIdx.x == 0) carry = 0u;
  __syncthreads();
  for (int base = 0; base < nbins; base += 1024) {
    unsigned v = hist[base + threadIdx.x];
    tmp[threadIdx.x] = v;
    __syncthreads();
    for (int off = 1; off < 1024; off <<= 1) {
      unsigned t = (threadIdx.x >= (unsigned)off) ? tmp[threadIdx.x - off] : 0u;
      __syncthreads();
      tmp[threadIdx.x] += t;
      __syncthreads();
    }
    pref[base + threadIdx.x] = carry + tmp[threadIdx.x] - v;  // exclusive
    __syncthreads();
    if (threadIdx.x == 1023) carry += tmp[1023];
    __syncthreads();
  }
}

__global__ void __launch_bounds__(256) k_scatter(const unsigned* __restrict__ key,
                                                 const unsigned* __restrict__ pref,
                                                 unsigned* __restrict__ cnt,
                                                 unsigned* __restrict__ skeys, int n) {
  int i = blockIdx.x * 256 + threadIdx.x;
  if (i >= n) return;
  unsigned k = key[i];
  unsigned b = k >> 15;
  unsigned p = pref[b] + atomicAdd(&cnt[b], 1u);
  skeys[p] = k;
}

__global__ void __launch_bounds__(256) k_rank(const unsigned* __restrict__ key,
                                              const unsigned* __restrict__ pref,
                                              const unsigned* __restrict__ hist,
                                              const unsigned* __restrict__ skeys,
                                              float* __restrict__ scores,
                                              int n, int accumulate) {
  int i = blockIdx.x * 256 + threadIdx.x;
  if (i >= n) return;
  unsigned k = key[i];
  unsigned b = k >> 15;
  unsigned base = pref[b], c = hist[b];
  unsigned less = 0, eq = 0;
  for (unsigned j = 0; j < c; ++j) {
    unsigned kk = skeys[base + j];
    less += (kk < k);
    eq   += (kk == k);
  }
  float rank = (float)(base + less) + 0.5f * (float)(eq - 1u);
  float s = rank * (1.0f / (float)n);
  float v = s * s;
  if (accumulate) scores[i] += v; else scores[i] = v;
}

// ---------------------------------------------------------------- top-K select
__global__ void __launch_bounds__(256) k_hist16(const unsigned* __restrict__ key,
                                                unsigned* __restrict__ hist, int n) {
  int i = blockIdx.x * 256 + threadIdx.x;
  if (i < n) atomicAdd(&hist[key[i] >> 16], 1u);
}

__global__ void k_findhi(const unsigned* __restrict__ hist, unsigned* scal) {
  unsigned cum = 0;
  for (int b = 65535; b >= 0; --b) {
    unsigned c = hist[b];
    if (cum + c >= KSEL) { scal[1] = (unsigned)b; scal[2] = cum; return; }
    cum += c;
  }
  scal[1] = 0; scal[2] = cum;
}

__global__ void __launch_bounds__(256) k_hist2(const unsigned* __restrict__ key,
                                               unsigned* __restrict__ hist2,
                                               const unsigned* __restrict__ scal, int n) {
  int i = blockIdx.x * 256 + threadIdx.x;
  if (i >= n) return;
  unsigned k = key[i];
  if ((k >> 16) == scal[1]) atomicAdd(&hist2[k & 0xFFFFu], 1u);
}

__global__ void k_findlo(const unsigned* __restrict__ hist2, unsigned* scal) {
  unsigned cum = scal[2];
  unsigned thi = scal[1];
  for (int lo = 65535; lo >= 0; --lo) {
    unsigned c = hist2[lo];
    if (cum + c >= KSEL) {
      scal[3] = (thi << 16) | (unsigned)lo;  // boundary key B
      scal[4] = cum;                         // G = #{key > B}
      break;
    }
    cum += c;
  }
  scal[5] = 0u;  // ctrA
  scal[6] = 0u;  // ctrB
}

__global__ void __launch_bounds__(256) k_select(const unsigned* __restrict__ key,
                                                unsigned* scal,
                                                unsigned long long* __restrict__ selk,
                                                int n) {
  int i = blockIdx.x * 256 + threadIdx.x;
  if (i >= n) return;
  unsigned k = key[i];
  unsigned B = scal[3], G = scal[4];
  if (k > B) {
    unsigned p = atomicAdd(&scal[5], 1u);
    selk[p] = ((unsigned long long)(~k) << 32) | (unsigned)i;
  } else if (k == B) {
    unsigned q = atomicAdd(&scal[6], 1u);
    if (G + q < KSEL)
      selk[G + q] = ((unsigned long long)(~k) << 32) | (unsigned)i;
  }
}

__global__ void __launch_bounds__(256) k_sortsel(const unsigned long long* __restrict__ selk,
                                                 int* __restrict__ idxsel) {
  __shared__ unsigned long long sv[KSEL];
  int t = threadIdx.x;
  sv[t] = selk[t];
  __syncthreads();
  for (int ksz = 2; ksz <= KSEL; ksz <<= 1)
    for (int j = ksz >> 1; j > 0; j >>= 1) {
      int ixj = t ^ j;
      if (ixj > t) {
        bool up = ((t & ksz) == 0);
        unsigned long long a = sv[t], b = sv[ixj];
        bool sw = up ? (a > b) : (a < b);
        if (sw) { sv[t] = b; sv[ixj] = a; }
      }
      __syncthreads();
    }
  idxsel[t] = (int)(sv[t] & 0xFFFFFFFFull);
}

// ---------------------------------------------------------------- WMMA GEMM
// C[M,N] = act(A[M,K] * W[N,K]^T + bias) (+ Res) * scale
// Block tile: 32 (M) x 64 (N), K-step 64.  128 threads = 4 waves; each wave
// owns two 16x16 accumulators (M-halves) sharing one B fragment.
// B panel DMA'd by TDM (tensor_load_to_lds) into a raw-f32 landing buffer,
// then converted block-wide to a packed-bf16 tile so the WMMA fragment build
// stays a pure ds_load_b128 -> v_wmma chain (no VALU on the critical path).
__global__ void __launch_bounds__(128)
k_gemm(const float* __restrict__ A, int lda,
       const float* __restrict__ W,
       const float* __restrict__ bias,
       const float* __restrict__ Res, int ldres,
       float* __restrict__ C, int ldc,
       int Mr, int N, int Kd, int act, float scale) {
  __shared__ unsigned short As[32][64];   // 4 KB
  __shared__ unsigned short Bs[64][64];   // 8 KB packed bf16
#ifdef GEMM_TDM
  __shared__ float Bsf[64][64];           // 16 KB raw f32 TDM landing
#endif
  const int tid  = threadIdx.x;
  const int wave = tid >> 5;
  const int lane = tid & 31;
  const int row0 = blockIdx.y * 32;
  const int col0 = blockIdx.x * 64;
  f32x8 acc0 = {};
  f32x8 acc1 = {};
  const int m  = lane & 15;
  const int ka = (lane >> 4) ? 8 : 0;    // A K-base per lane half
  const int nb = (lane & 15) + (wave << 4);
  const int kb = (lane >> 4) ? 8 : 0;    // B K-base per lane half
  const float4 zero4 = {0.f, 0.f, 0.f, 0.f};
#ifdef GEMM_TDM
  const unsigned ldsB = (unsigned)(unsigned long long)(const void*)&Bsf[0][0];
#endif
  for (int k0 = 0; k0 < Kd; k0 += 64) {
#ifdef GEMM_TDM
    if (tid < 32) {
      // D# group0: count=1 | lds_addr | global_addr[56:0] | type=2
      unsigned long long ga =
          (unsigned long long)(const void*)(W + (size_t)col0 * Kd + k0);
      u32x4v g0;
      g0[0] = 1u;                                           // count=1
      g0[1] = ldsB;                                         // lds_addr (bytes)
      g0[2] = (unsigned)ga;                                 // global_addr lo
      g0[3] = ((unsigned)(ga >> 32) & 0x01FFFFFFu) | 0x80000000u; // hi | type=2
      // D# group1: mask=0, data_size=4B, dims/strides for a 64x64 f32 tile
      i32x8v g1 = {};
      g1[0] = 2 << 16;                    // data_size = 2 (4 bytes)
      g1[1] = (Kd & 0xFFFF) << 16;        // tensor_dim0[15:0]
      g1[2] = (N & 0xFFFF) << 16;         // tensor_dim1[15:0] (dim0 hi = 0)
      g1[3] = 64 << 16;                   // tile_dim0 = 64 (dim1 hi = 0)
      g1[4] = 64;                         // tile_dim1 = 64
      g1[5] = Kd;                         // tensor_dim0_stride[31:0]
      i32x4v gz4 = {};
#if __clang_major__ >= 23
      i32x8v gz8 = {};
      __builtin_amdgcn_tensor_load_to_lds(g0, g1, gz4, gz4, gz8, 0);
#else
      __builtin_amdgcn_tensor_load_to_lds(g0, g1, gz4, gz4, 0);
#endif
    }
#endif
    // stage A: 32x64 f32 = 512 float4, 4 per thread (overlaps B DMA)
#pragma unroll
    for (int i = 0; i < 4; ++i) {
      int e = tid + i * 128;
      int r = e >> 4, c4 = e & 15;
      int gr = row0 + r;
      float4 v = (gr < Mr)
                   ? *(const float4*)(A + (size_t)gr * lda + k0 + c4 * 4)
                   : zero4;
      *(uint2*)&As[r][c4 * 4] = pack4bf(v);
    }
#ifdef GEMM_TDM
    if (tid < 32) __builtin_amdgcn_s_wait_tensorcnt(0);
    __syncthreads();
    // block-wide convert: raw f32 TDM tile -> packed bf16 tile (off the
    // WMMA critical path)
#pragma unroll
    for (int i = 0; i < 8; ++i) {
      int e = tid + i * 128;
      int r = e >> 4, c4 = e & 15;
      float4 v = *(const float4*)&Bsf[r][c4 * 4];
      *(uint2*)&Bs[r][c4 * 4] = pack4bf(v);
    }
#else
    // stage B: 64x64 f32 = 1024 float4, 8 per thread (N multiple of 64)
#pragma unroll
    for (int i = 0; i < 8; ++i) {
      int e = tid + i * 128;
      int r = e >> 4, c4 = e & 15;
      float4 v = *(const float4*)(W + (size_t)(col0 + r) * Kd + k0 + c4 * 4);
      *(uint2*)&Bs[r][c4 * 4] = pack4bf(v);
    }
#endif
    __syncthreads();
#pragma unroll
    for (int kh = 0; kh < 2; ++kh) {
      u16x16 au0, au1, bu;
      const int kof = kh * 32;
#pragma unroll
      for (int e = 0; e < 16; ++e) {
        int kk = kof + ka + e + ((e < 8) ? 0 : 8);
        au0[e] = As[m][kk];
        au1[e] = As[16 + m][kk];
        bu[e]  = Bs[nb][kof + kb + e + ((e < 8) ? 0 : 8)];
      }
      bf16x16 a0 = __builtin_bit_cast(bf16x16, au0);
      bf16x16 a1 = __builtin_bit_cast(bf16x16, au1);
      bf16x16 bf = __builtin_bit_cast(bf16x16, bu);
      acc0 = __builtin_amdgcn_wmma_f32_16x16x32_bf16(false, a0, false, bf,
                                                     (short)0, acc0, false, false);
      acc1 = __builtin_amdgcn_wmma_f32_16x16x32_bf16(false, a1, false, bf,
                                                     (short)0, acc1, false, false);
    }
    __syncthreads();
  }
  int gcol = col0 + nb;
  float bv = bias ? bias[gcol] : 0.f;
#pragma unroll
  for (int r = 0; r < 8; ++r) {
    int grow = row0 + ((lane >> 4) ? (8 + r) : r);
    if (grow < Mr) {
      float v = acc0[r] + bv;
      if (act == 1) v = gelu_f(v);
      if (Res) v += Res[(size_t)grow * ldres + gcol];
      C[(size_t)grow * ldc + gcol] = v * scale;
    }
    int grow1 = grow + 16;
    if (grow1 < Mr) {
      float v = acc1[r] + bv;
      if (act == 1) v = gelu_f(v);
      if (Res) v += Res[(size_t)grow1 * ldres + gcol];
      C[(size_t)grow1 * ldc + gcol] = v * scale;
    }
  }
}

// ---------------------------------------------------------------- LayerNorm
__global__ void __launch_bounds__(256) k_ln(const float* __restrict__ in,
                                            float* __restrict__ out,
                                            const float* __restrict__ g,
                                            const float* __restrict__ b) {
  int row = blockIdx.x;
  __shared__ float red[9];
  const float* r = in + (size_t)row * D_MODEL;
  float v[4];
  float s = 0.f;
#pragma unroll
  for (int i = 0; i < 4; ++i) { v[i] = r[threadIdx.x + i * 256]; s += v[i]; }
#pragma unroll
  for (int o = 16; o > 0; o >>= 1) s += __shfl_xor(s, o, 32);
  if ((threadIdx.x & 31) == 0) red[threadIdx.x >> 5] = s;
  __syncthreads();
  if (threadIdx.x == 0) {
    float t = 0.f; for (int i = 0; i < 8; ++i) t += red[i];
    red[8] = t * (1.0f / D_MODEL);
  }
  __syncthreads();
  float mean = red[8];
  float s2 = 0.f;
#pragma unroll
  for (int i = 0; i < 4; ++i) { float d = v[i] - mean; s2 += d * d; }
#pragma unroll
  for (int o = 16; o > 0; o >>= 1) s2 += __shfl_xor(s2, o, 32);
  __syncthreads();
  if ((threadIdx.x & 31) == 0) red[threadIdx.x >> 5] = s2;
  __syncthreads();
  if (threadIdx.x == 0) {
    float t = 0.f; for (int i = 0; i < 8; ++i) t += red[i];
    red[8] = rsqrtf(t * (1.0f / D_MODEL) + 1e-5f);
  }
  __syncthreads();
  float rs = red[8];
#pragma unroll
  for (int i = 0; i < 4; ++i) {
    int c = threadIdx.x + i * 256;
    out[(size_t)row * D_MODEL + c] = (v[i] - mean) * rs * g[c] + b[c];
  }
}

// ---------------------------------------------------------------- attention
// Causal: query tile [q0, q0+16) only needs keys kj < min(SEQ, q0+16).
__global__ void __launch_bounds__(256) k_attn(const float* __restrict__ qkv,
                                              float* __restrict__ out) {
  const int h  = blockIdx.y;
  const int q0 = blockIdx.x * 16;
  const int kjmax = min(SEQ, q0 + 16);
  __shared__ float sc[16][516];
  __shared__ float qt[16][64];
  __shared__ float rr[16][17];
  const int tid = threadIdx.x;
  for (int e = tid; e < 16 * 64; e += 256) {
    int qi = e >> 6, d = e & 63;
    int gq = q0 + qi;
    qt[qi][d] = (gq < SEQ) ? qkv[(size_t)gq * 3072 + h * DHEAD + d] : 0.f;
  }
  __syncthreads();
  for (int e = tid; e < 16 * kjmax; e += 256) {
    int qi = e / kjmax, kj = e - qi * kjmax;
    int gq = q0 + qi;
    const float* kv = qkv + (size_t)kj * 3072 + D_MODEL + h * DHEAD;
    float a = 0.f;
#pragma unroll 8
    for (int d = 0; d < DHEAD; ++d) a += qt[qi][d] * kv[d];
    a *= 0.125f;                         // 1/sqrt(64)
    if (kj > gq) a = -1e9f;
    sc[qi][kj] = a;
  }
  __syncthreads();
  const int row = tid >> 4, sub = tid & 15;
  float mx = -3.0e38f;
  for (int j = sub; j < kjmax; j += 16) mx = fmaxf(mx, sc[row][j]);
  rr[row][sub] = mx;
  __syncthreads();
  if (sub == 0) {
    float m = rr[row][0];
    for (int i = 1; i < 16; ++i) m = fmaxf(m, rr[row][i]);
    rr[row][16] = m;
  }
  __syncthreads();
  float m = rr[row][16];
  float sum = 0.f;
  for (int j = sub; j < kjmax; j += 16) {
    float ex = __expf(sc[row][j] - m);
    sc[row][j] = ex;
    sum += ex;
  }
  __syncthreads();
  rr[row][sub] = sum;
  __syncthreads();
  if (sub == 0) {
    float t = 0.f;
    for (int i = 0; i < 16; ++i) t += rr[row][i];
    rr[row][16] = 1.0f / t;
  }
  __syncthreads();
  for (int e = tid; e < 16 * 64; e += 256) {
    int qi = e >> 6, d = e & 63;
    int gq = q0 + qi;
    if (gq >= SEQ) continue;
    float a = 0.f;
    const float* vbase = qkv + 2 * D_MODEL + h * DHEAD + d;
    for (int kj = 0; kj <= gq; ++kj)
      a += sc[qi][kj] * vbase[(size_t)kj * 3072];
    out[(size_t)gq * D_MODEL + h * DHEAD + d] = a * rr[qi][16];
  }
}

// ---------------------------------------------------------------- small ops
__global__ void __launch_bounds__(256) k_seq(const float* __restrict__ stok,
                                             const float* __restrict__ xv,
                                             const float* __restrict__ pemb,
                                             const int* __restrict__ idxsel,
                                             const float* __restrict__ mv,
                                             const float* __restrict__ avec,
                                             float* __restrict__ vec) {
  int s = blockIdx.x;
  int d = blockIdx.y * 256 + threadIdx.x;
  float base;
  if (s == 0) base = stok[d];
  else if (s == SEQ - 1) base = xv[d];
  else {
    int k = (s - 1) >> 1;
    if (((s - 1) & 1) == 0) base = mv[(size_t)idxsel[k] * D_MODEL + d];
    else                    base = avec[(size_t)k * D_MODEL + d];
  }
  vec[(size_t)s * D_MODEL + d] = base + pemb[(size_t)s * D_MODEL + d];
}

__global__ void __launch_bounds__(256) k_actpre(const int* __restrict__ aidx,
                                                const int* __restrict__ idxsel,
                                                const float* __restrict__ w1,
                                                const float* __restrict__ b1,
                                                float* __restrict__ out, int width) {
  int k = blockIdx.x;
  int d = blockIdx.y * 256 + threadIdx.x;
  int a = aidx[idxsel[k]];
  float v = w1[(size_t)d * width + a] + b1[d];
  out[(size_t)k * D_MODEL + d] = gelu_f(v);
}

__global__ void __launch_bounds__(256) k_vhgather(const float* __restrict__ vec,
                                                  float* __restrict__ vhin) {
  int j = blockIdx.x;
  int d = blockIdx.y * 256 + threadIdx.x;
  vhin[(size_t)j * D_MODEL + d] = vec[(size_t)(2 * j + 1) * D_MODEL + d];
}

__global__ void __launch_bounds__(256) k_vh2(const float* __restrict__ hid,
                                             const float* __restrict__ w2,
                                             const float* __restrict__ b2,
                                             float* __restrict__ vpred) {
  int row = blockIdx.x;
  __shared__ float red[8];
  float s = 0.f;
  for (int i = threadIdx.x; i < 4096; i += 256)
    s += hid[(size_t)row * 4096 + i] * w2[i];
#pragma unroll
  for (int o = 16; o > 0; o >>= 1) s += __shfl_xor(s, o, 32);
  if ((threadIdx.x & 31) == 0) red[threadIdx.x >> 5] = s;
  __syncthreads();
  if (threadIdx.x == 0) {
    float t = 0.f; for (int i = 0; i < 8; ++i) t += red[i];
    vpred[row] = t + b2[0];
  }
}

__global__ void __launch_bounds__(256) k_tail(const float* __restrict__ vpred,
                                              const int* __restrict__ idxsel,
                                              const float* __restrict__ mrew,
                                              float* __restrict__ out,
                                              int off_val, int off_loss, int off_rew) {
  int k = threadIdx.x;
  float r = mrew[idxsel[k]];
  float d = vpred[k] - r;
  out[off_loss + k] = d * d;
  out[off_rew + k]  = r;
  if (k == 0) out[off_val] = vpred[KSEL];
}

__global__ void __launch_bounds__(256) k_copyvec(const float* __restrict__ vec,
                                                 float* __restrict__ out) {
  int s = blockIdx.x;
  int d = blockIdx.y * 256 + threadIdx.x;
  out[(size_t)s * D_MODEL + d] = vec[(size_t)(s + 1) * D_MODEL + d];
}

__global__ void __launch_bounds__(256) k_gnext(const float* __restrict__ mnext,
                                               const int* __restrict__ idxsel,
                                               float* __restrict__ out) {
  int k = blockIdx.x;
  int d = blockIdx.y * 256 + threadIdx.x;
  out[(size_t)k * D_MODEL + d] = mnext[(size_t)idxsel[k] * D_MODEL + d];
}

// ---------------------------------------------------------------- launcher
extern "C" void kernel_launch(void* const* d_in, const int* in_sizes, int n_in,
                              void* d_out, int out_size, void* d_ws, size_t ws_size,
                              hipStream_t stream) {
  const float* xv    = (const float*)d_in[0];
  const float* mv    = (const float*)d_in[1];
  const float* mnext = (const float*)d_in[2];
  const float* msur  = (const float*)d_in[3];
  const float* mrew  = (const float*)d_in[4];
  const float* stoch = (const float*)d_in[5];
  const int*   a0i   = (const int*)d_in[6];
  const int*   a1i   = (const int*)d_in[7];
  const float* stok  = (const float*)d_in[8];
  const float* pemb  = (const float*)d_in[9];
  const float* ln1g  = (const float*)d_in[10];
  const float* ln1b  = (const float*)d_in[11];
  const float* wqkv  = (const float*)d_in[12];
  const float* bqkv  = (const float*)d_in[13];
  const float* wo    = (const float*)d_in[14];
  const float* bo    = (const float*)d_in[15];
  const float* ln2g  = (const float*)d_in[16];
  const float* ln2b  = (const float*)d_in[17];
  const float* w1    = (const float*)d_in[18];
  const float* b1    = (const float*)d_in[19];
  const float* w2    = (const float*)d_in[20];
  const float* b2    = (const float*)d_in[21];
  const float* a0w1  = (const float*)d_in[22];
  const float* a0b1  = (const float*)d_in[23];
  const float* a0w2  = (const float*)d_in[24];
  const float* a0b2  = (const float*)d_in[25];
  const float* a1w1  = (const float*)d_in[26];
  const float* a1b1  = (const float*)d_in[27];
  const float* a1w2  = (const float*)d_in[28];
  const float* a1b2  = (const float*)d_in[29];
  const float* vhw1  = (const float*)d_in[30];
  const float* vhb1  = (const float*)d_in[31];
  const float* vhw2  = (const float*)d_in[32];
  const float* vhb2  = (const float*)d_in[33];
  float* out = (float*)d_out;

  size_t off = 0;
  auto carve = [&](size_t bytes) -> char* {
    char* p = (char*)d_ws + off;
    off += (bytes + 1023) & ~(size_t)1023;
    return p;
  };
  float*    sims   = (float*)carve((size_t)MMEM * 4);
  float*    scores = (float*)carve((size_t)MMEM * 4);
  unsigned* key    = (unsigned*)carve((size_t)MMEM * 4);
  unsigned* skeys  = (unsigned*)carve((size_t)MMEM * 4);
  unsigned* hist   = (unsigned*)carve((size_t)RBINS * 4);
  unsigned* pref   = (unsigned*)carve((size_t)RBINS * 4);
  unsigned* cnt    = (unsigned*)carve((size_t)RBINS * 4);
  unsigned* scal   = (unsigned*)carve(4096);
  unsigned long long* selk = (unsigned long long*)carve(KSEL * 8);
  int*      idxsel = (int*)carve(KSEL * 4);
  float*    vec    = (float*)carve((size_t)SEQ * D_MODEL * 4);
  float*    hbuf   = (float*)carve((size_t)SEQ * D_MODEL * 4);
  float*    qkvb   = (float*)carve((size_t)SEQ * 3 * D_MODEL * 4);
  float*    attb   = (float*)carve((size_t)SEQ * D_MODEL * 4);
  float*    mlph   = (float*)carve((size_t)SEQ * 4 * D_MODEL * 4);
  float*    apre   = (float*)carve((size_t)KSEL * D_MODEL * 4);
  float*    h0b    = (float*)carve((size_t)KSEL * D_MODEL * 4);
  float*    avec   = (float*)carve((size_t)KSEL * D_MODEL * 4);
  float*    vhin   = (float*)carve((size_t)(KSEL + 1) * D_MODEL * 4);
  float*    vhhid  = (float*)carve((size_t)(KSEL + 1) * 4 * D_MODEL * 4);
  float*    vpred  = (float*)carve((size_t)(KSEL + 1) * 4);
  float*    scalf  = (float*)scal;

  const int MB = (MMEM + 255) / 256;
  const int ZB = (RBINS + 255) / 256;

  // --- retrieval ---
  k_xnorm<<<1, 256, 0, stream>>>(xv, scalf);
  k_sims<<<(MMEM + 7) / 8, 256, 0, stream>>>(mv, xv, scalf, sims);

  // --- three exact tie-averaged rank passes ---
  const float* rank_src[3] = {msur, sims, stoch};
  for (int p = 0; p < 3; ++p) {
    k_zero_u32<<<ZB, 256, 0, stream>>>(hist, RBINS);
    k_zero_u32<<<ZB, 256, 0, stream>>>(cnt, RBINS);
    k_keys<<<MB, 256, 0, stream>>>(rank_src[p], key, MMEM);
    k_hist15<<<MB, 256, 0, stream>>>(key, hist, MMEM);
    k_prefix<<<1, 1024, 0, stream>>>(hist, pref, RBINS);
    k_scatter<<<MB, 256, 0, stream>>>(key, pref, cnt, skeys, MMEM);
    k_rank<<<MB, 256, 0, stream>>>(key, pref, hist, skeys, scores, MMEM, p > 0);
  }

  // --- top-K=256 radix select + descending sort ---
  k_zero_u32<<<ZB, 256, 0, stream>>>(hist, RBINS);  // hist16 + hist2 regions
  k_keys<<<MB, 256, 0, stream>>>(scores, key, MMEM);
  k_hist16<<<MB, 256, 0, stream>>>(key, hist, MMEM);
  k_findhi<<<1, 1, 0, stream>>>(hist, scal);
  k_hist2<<<MB, 256, 0, stream>>>(key, hist + 65536, scal, MMEM);
  k_findlo<<<1, 1, 0, stream>>>(hist + 65536, scal);
  k_select<<<MB, 256, 0, stream>>>(key, scal, selk, MMEM);
  k_sortsel<<<1, KSEL, 0, stream>>>(selk, idxsel);

  // --- action encoders: (h0 + h1) * 0.5 ---
  k_actpre<<<dim3(KSEL, 4), 256, 0, stream>>>(a0i, idxsel, a0w1, a0b1, apre, 16);
  k_gemm<<<dim3(16, 8), 128, 0, stream>>>(apre, D_MODEL, a0w2, a0b2,
                                          nullptr, 0, h0b, D_MODEL,
                                          KSEL, D_MODEL, D_MODEL, 0, 1.0f);
  k_actpre<<<dim3(KSEL, 4), 256, 0, stream>>>(a1i, idxsel, a1w1, a1b1, apre, 8);
  k_gemm<<<dim3(16, 8), 128, 0, stream>>>(apre, D_MODEL, a1w2, a1b2,
                                          h0b, D_MODEL, avec, D_MODEL,
                                          KSEL, D_MODEL, D_MODEL, 0, 0.5f);

  // --- build sequence ---
  k_seq<<<dim3(SEQ, 4), 256, 0, stream>>>(stok, xv, pemb, idxsel, mv, avec, vec);

  // --- transformer layers ---
  const int MT16 = (SEQ + 15) / 16;  // 33 (attention q-tiles)
  const int MT32 = (SEQ + 31) / 32;  // 17 (gemm row-tiles)
  for (int i = 0; i < NLAYER; ++i) {
    const float* wq = wqkv + (size_t)i * 3 * D_MODEL * D_MODEL;
    const float* bq = bqkv + (size_t)i * 3 * D_MODEL;
    const float* wp = wo + (size_t)i * D_MODEL * D_MODEL;
    const float* bp = bo + (size_t)i * D_MODEL;
    const float* wm1 = w1 + (size_t)i * 4 * D_MODEL * D_MODEL;
    const float* bm1 = b1 + (size_t)i * 4 * D_MODEL;
    const float* wm2 = w2 + (size_t)i * D_MODEL * 4 * D_MODEL;
    const float* bm2 = b2 + (size_t)i * D_MODEL;

    k_ln<<<SEQ, 256, 0, stream>>>(vec, hbuf, ln1g + (size_t)i * D_MODEL,
                                  ln1b + (size_t)i * D_MODEL);
    k_gemm<<<dim3(48, MT32), 128, 0, stream>>>(hbuf, D_MODEL, wq, bq,
                                               nullptr, 0, qkvb, 3 * D_MODEL,
                                               SEQ, 3 * D_MODEL, D_MODEL, 0, 1.0f);
    k_attn<<<dim3(MT16, NHEAD), 256, 0, stream>>>(qkvb, attb);
    k_gemm<<<dim3(16, MT32), 128, 0, stream>>>(attb, D_MODEL, wp, bp,
                                               vec, D_MODEL, vec, D_MODEL,
                                               SEQ, D_MODEL, D_MODEL, 0, 1.0f);
    k_ln<<<SEQ, 256, 0, stream>>>(vec, hbuf, ln2g + (size_t)i * D_MODEL,
                                  ln2b + (size_t)i * D_MODEL);
    k_gemm<<<dim3(64, MT32), 128, 0, stream>>>(hbuf, D_MODEL, wm1, bm1,
                                               nullptr, 0, mlph, 4 * D_MODEL,
                                               SEQ, 4 * D_MODEL, D_MODEL, 1, 1.0f);
    k_gemm<<<dim3(16, MT32), 128, 0, stream>>>(mlph, 4 * D_MODEL, wm2, bm2,
                                               vec, D_MODEL, vec, D_MODEL,
                                               SEQ, D_MODEL, 4 * D_MODEL, 0, 1.0f);
  }

  // --- value head ---
  k_vhgather<<<dim3(KSEL + 1, 4), 256, 0, stream>>>(vec, vhin);
  k_gemm<<<dim3(64, (KSEL + 1 + 31) / 32), 128, 0, stream>>>(
      vhin, D_MODEL, vhw1, vhb1, nullptr, 0, vhhid, 4 * D_MODEL,
      KSEL + 1, 4 * D_MODEL, D_MODEL, 1, 1.0f);
  k_vh2<<<KSEL + 1, 256, 0, stream>>>(vhhid, vhw2, vhb2, vpred);

  // --- outputs ---
  const int OUT_VEC  = 0;
  const int OUT_NEXT = (SEQ - 1) * D_MODEL;            // 525312
  const int OUT_VAL  = OUT_NEXT + KSEL * D_MODEL;      // 787456
  const int OUT_LOSS = OUT_VAL + 1;                    // 787457
  const int OUT_REW  = OUT_LOSS + KSEL;                // 787713
  k_copyvec<<<dim3(SEQ - 1, 4), 256, 0, stream>>>(vec, out + OUT_VEC);
  k_gnext<<<dim3(KSEL, 4), 256, 0, stream>>>(mnext, idxsel, out + OUT_NEXT);
  k_tail<<<1, KSEL, 0, stream>>>(vpred, idxsel, mrew, out, OUT_VAL, OUT_LOSS, OUT_REW);
}
